// WindowAttention3D_78735340470549
// MI455X (gfx1250) — matrix operand order repaired and checked by reading, not compiled, not run on hardware
//
#include <hip/hip_runtime.h>

typedef __attribute__((ext_vector_type(16))) _Float16 v16h;
typedef __attribute__((ext_vector_type(8)))  _Float16 v8h;
typedef __attribute__((ext_vector_type(8)))  float    v8f;

#define N_TOK 98
#define NP 112
#define CDIM 192
#define NHEAD 6
#define HDIM 32
#define NWIN 512
#define NWINDOWS 2048
#define QK_SCALE 0.17677669529663687f   // 1/sqrt(32)

// ---------------------------------------------------------------------------
// WMMA fragment helpers (CDNA5 16x16x32 f16, wave32).
// A-matrix per-lane layout (ISA 7.12.2, 16-bit A 16x32):
//   lane L<16  -> row M=L,   K = {k0+0..7,  k0+16..23}
//   lane L>=16 -> row M=L-16,K = {k0+8..15, k0+24..31}
// B-matrix per-lane layout (32x16):
//   lane L<16  -> col N=L,   K = {k0+0..7,  k0+16..23}
//   lane L>=16 -> col N=L-16,K = {k0+8..15, k0+24..31}
// ---------------------------------------------------------------------------

// bank-swizzle for stride-192 (384B) f16 arrays: permute 16B chunks within
// aligned groups of 8 so the 16 lanes of a fragment load hit 16 disjoint
// 4-bank spans.  swz(r) distinct per parity class over r&15.
__device__ __forceinline__ int swz8(int r) { return ((r & 7) + ((r >> 3) & 1)) & 7; }
__device__ __forceinline__ int swz_idx(int row, int c) {
  return row * CDIM + ((((c >> 3) ^ swz8(row)) << 3) | (c & 7));
}

// A-fragment from swizzled stride-192 LDS array (Xs / Os)
__device__ __forceinline__ v16h frag_a_swz(const _Float16* base, int row0, int k0) {
  const int lane = threadIdx.x & 31;
  const int r  = row0 + (lane & 15);
  const int sw = swz8(r);
  const int ch = (k0 >> 3) + ((lane & 16) ? 1 : 0);
  const _Float16* pr = base + r * CDIM;
  v8h lo = *(const v8h*)(pr + (((ch    ) ^ sw) << 3));
  v8h hi = *(const v8h*)(pr + (((ch + 2) ^ sw) << 3));
  return __builtin_shufflevector(lo, hi, 0,1,2,3,4,5,6,7,8,9,10,11,12,13,14,15);
}

// A-fragment from linear storage (Q, P)
__device__ __forceinline__ v16h frag_a(const _Float16* base, int row0, int k0, int stride) {
  const int lane = threadIdx.x & 31;
  const _Float16* p = base + (row0 + (lane & 15)) * stride + k0 + ((lane & 16) ? 8 : 0);
  v8h lo = *(const v8h*)p;
  v8h hi = *(const v8h*)(p + 16);
  return __builtin_shufflevector(lo, hi, 0,1,2,3,4,5,6,7,8,9,10,11,12,13,14,15);
}

// B-fragment from [col][k] row-major storage (weights, K, Vt)
__device__ __forceinline__ v16h frag_b(const _Float16* base, int col0, int k0, int stride) {
  const int lane = threadIdx.x & 31;
  const _Float16* p = base + (col0 + (lane & 15)) * stride + k0 + ((lane & 16) ? 8 : 0);
  v8h lo = *(const v8h*)p;
  v8h hi = *(const v8h*)(p + 16);
  return __builtin_shufflevector(lo, hi, 0,1,2,3,4,5,6,7,8,9,10,11,12,13,14,15);
}

__device__ __forceinline__ v8f wmma_f16(v16h a, v16h b, v8f c) {
  return __builtin_amdgcn_wmma_f32_16x16x32_f16(false, a, false, b, (short)0, c, false, false);
}

// ---------------------------------------------------------------------------
// Prep: f32->f16 weight conversion + relative-position-bias gather.
// ws layout: w3h (3C*C f16), pwh (C*C f16), biasF (H*N*N f32)
// ---------------------------------------------------------------------------
__global__ void prep_kernel(const float* __restrict__ qkv_w,
                            const float* __restrict__ proj_w,
                            const float* __restrict__ rpb,
                            const int*   __restrict__ rpi,
                            _Float16* __restrict__ w3h,
                            _Float16* __restrict__ pwh,
                            float*    __restrict__ biasF) {
  int idx = blockIdx.x * blockDim.x + threadIdx.x;
  const int n_w3   = 3 * CDIM * CDIM;            // 110592
  const int n_pw   = CDIM * CDIM;                // 36864
  const int n_bias = NHEAD * N_TOK * N_TOK;      // 57624
  if (idx < n_w3) { w3h[idx] = (_Float16)qkv_w[idx]; return; }
  idx -= n_w3;
  if (idx < n_pw) { pwh[idx] = (_Float16)proj_w[idx]; return; }
  idx -= n_pw;
  if (idx < n_bias) {
    int h = idx / (N_TOK * N_TOK);
    int p = idx % (N_TOK * N_TOK);
    biasF[idx] = rpb[rpi[p] * NHEAD + h];
  }
}

// ---------------------------------------------------------------------------
// Fused window attention: one block = one window (98 tokens, 6 heads).
// ---------------------------------------------------------------------------
__global__ __launch_bounds__(256)
void attn_kernel(const float* __restrict__ x,
                 const float* __restrict__ mask,
                 const float* __restrict__ qkv_b,
                 const float* __restrict__ proj_b,
                 const _Float16* __restrict__ w3h,
                 const _Float16* __restrict__ pwh,
                 const float* __restrict__ biasF,
                 float* __restrict__ out) {
  __shared__ __attribute__((aligned(16))) _Float16 Xs [NP * CDIM]; // 43008 B (swizzled)
  __shared__ __attribute__((aligned(16))) float    Ssm[NP * NP];   // 50176 B (P overlays as f16)
  __shared__ __attribute__((aligned(16))) _Float16 Qh [NP * 40];   //  8960 B
  __shared__ __attribute__((aligned(16))) _Float16 Kh [NP * 40];   //  8960 B
  __shared__ __attribute__((aligned(16))) _Float16 Vth[HDIM * 136];//  8704 B ([d][token])
  __shared__ __attribute__((aligned(16))) _Float16 Os [NP * CDIM]; // 43008 B (swizzled)
  // total 162816 B (< 160 KB)

  const int b    = blockIdx.x;
  const int tid  = threadIdx.x;
  const int wave = tid >> 5;
  const int lane = tid & 31;
  const int lc   = lane & 15;
  const int rsel = (lane & 16) ? 8 : 0;

  // ---- stage X (f32 -> f16, swizzled), zero pad rows, zero Vt ----
  const float* xb = x + (size_t)b * (N_TOK * CDIM);
  for (int i = tid; i < N_TOK * CDIM; i += 256)
    Xs[swz_idx(i / CDIM, i % CDIM)] = (_Float16)xb[i];
  for (int i = tid; i < (NP - N_TOK) * CDIM; i += 256)
    Xs[N_TOK * CDIM + i] = (_Float16)0.f;      // whole rows: swizzle-invariant
  for (int i = tid; i < (HDIM * 136) / 2; i += 256)
    ((unsigned int*)Vth)[i] = 0u;
  __syncthreads();

  const float* maskb = mask + (size_t)(b % NWIN) * (N_TOK * N_TOK);
  float* outb = out + (size_t)b * (N_TOK * CDIM);

  for (int h = 0; h < NHEAD; ++h) {
    // ---- Phase 1: per-head QKV GEMM. 7 m-tiles x (3 mats x 2 d-halves) ----
    for (int t = wave; t < 42; t += 8) {
      int mt = t / 6;
      int ct = t % 6;
      int s  = __builtin_amdgcn_readfirstlane(ct >> 1);   // 0=q 1=k 2=v (wave-uniform)
      int d0 = (ct & 1) * 16;
      int col0 = (ct >> 1) * CDIM + h * HDIM + d0;
      v8f acc = {};
#pragma unroll
      for (int ks = 0; ks < 6; ++ks) {
        v16h a  = frag_a_swz(Xs, mt * 16, ks * 32);
        v16h bf = frag_b(w3h, col0, ks * 32, CDIM);
        acc = wmma_f16(a, bf, acc);
      }
      float bias = qkv_b[col0 + lc];
      int d = d0 + lc;
      int rbase = mt * 16 + rsel;
      if (s == 0) {
#pragma unroll
        for (int i = 0; i < 8; ++i)
          Qh[(rbase + i) * 40 + d] = (_Float16)((acc[i] + bias) * QK_SCALE);
      } else if (s == 1) {
#pragma unroll
        for (int i = 0; i < 8; ++i)
          Kh[(rbase + i) * 40 + d] = (_Float16)(acc[i] + bias);
      } else {
#pragma unroll
        for (int i = 0; i < 8; ++i)
          Vth[d * 136 + rbase + i] = (_Float16)(acc[i] + bias);
      }
    }
    __syncthreads();

    // ---- Phase 2a: S = (scaled Q) K^T + bias + mask  (7x7 tiles, K=32) ----
    const float* biash = biasF + h * (N_TOK * N_TOK);
    for (int t = wave; t < 49; t += 8) {
      int it = t / 7, jt = t % 7;
      v16h a  = frag_a(Qh, it * 16, 0, 40);
      v16h bf = frag_b(Kh, jt * 16, 0, 40);
      v8f acc = {};
      acc = wmma_f16(a, bf, acc);
      int col = jt * 16 + lc;
      int cc = min(col, N_TOK - 1);
#pragma unroll
      for (int i = 0; i < 8; ++i) {
        int row = it * 16 + i + rsel;
        int rc = min(row, N_TOK - 1);
        float add = biash[rc * N_TOK + cc] + maskb[rc * N_TOK + cc];
        float v = (row < N_TOK && col < N_TOK) ? acc[i] + add : 0.f;
        Ssm[row * NP + col] = v;
      }
    }
    __syncthreads();

    // ---- Phase 2b: softmax, wave-per-row, shuffle reductions.           ----
    // All S reads feed the reductions, so P stores (in-place f16 overlay)
    // are data-dependent on every read: no read/write hazard.
    {
      _Float16* Pbase = (_Float16*)Ssm;        // row stride 224 halves
      for (int r = wave; r < NP; r += 8) {
        _Float16* Pr = Pbase + r * (NP * 2);
        if (r < N_TOK) {
          const float* Sr = Ssm + r * NP;
          float a0 = Sr[lane];
          float a1 = Sr[lane + 32];
          float a2 = Sr[lane + 64];
          float a3 = (lane + 96 < N_TOK) ? Sr[lane + 96] : -1e30f;
          float m = fmaxf(fmaxf(a0, a1), fmaxf(a2, a3));
#pragma unroll
          for (int off = 16; off > 0; off >>= 1)
            m = fmaxf(m, __shfl_xor(m, off, 32));
          float e0 = __expf(a0 - m);
          float e1 = __expf(a1 - m);
          float e2 = __expf(a2 - m);
          float e3 = (lane + 96 < N_TOK) ? __expf(a3 - m) : 0.f;
          float s = e0 + e1 + e2 + e3;
#pragma unroll
          for (int off = 16; off > 0; off >>= 1)
            s += __shfl_xor(s, off, 32);
          float inv = 1.f / s;
          Pr[lane]      = (_Float16)(e0 * inv);
          Pr[lane + 32] = (_Float16)(e1 * inv);
          Pr[lane + 64] = (_Float16)(e2 * inv);
          Pr[lane + 96] = (_Float16)(e3 * inv);
        } else {
          Pr[lane]      = (_Float16)0.f;
          Pr[lane + 32] = (_Float16)0.f;
          Pr[lane + 64] = (_Float16)0.f;
          Pr[lane + 96] = (_Float16)0.f;
        }
      }
    }
    __syncthreads();

    // ---- Phase 2c: O_h = P @ V  (7 m-tiles x 2 d-tiles, K padded to 128) ----
    {
      const _Float16* P = (const _Float16*)Ssm;
      for (int t = wave; t < 14; t += 8) {
        int mt = t >> 1, dt = t & 1;
        v8f acc = {};
#pragma unroll
        for (int ks = 0; ks < 4; ++ks) {
          v16h a  = frag_a(P, mt * 16, ks * 32, NP * 2);
          v16h bf = frag_b(Vth, dt * 16, ks * 32, 136);
          acc = wmma_f16(a, bf, acc);
        }
        int c = h * HDIM + dt * 16 + lc;
#pragma unroll
        for (int i = 0; i < 8; ++i) {
          int row = mt * 16 + i + rsel;
          Os[swz_idx(row, c)] = (_Float16)acc[i];
        }
      }
    }
    __syncthreads();   // Qh/Kh/Vth/Ssm reused by next head
  }

  // ---- Phase 3: out = O @ proj_w^T + proj_b  (7 x 12 tiles, K=192) ----
  for (int t = wave; t < 84; t += 8) {
    int mt = t / 12, nt = t % 12;
    v8f acc = {};
#pragma unroll
    for (int ks = 0; ks < 6; ++ks) {
      v16h a  = frag_a_swz(Os, mt * 16, ks * 32);
      v16h bf = frag_b(pwh, nt * 16, ks * 32, CDIM);
      acc = wmma_f16(a, bf, acc);
    }
    float pb = proj_b[nt * 16 + lc];
#pragma unroll
    for (int i = 0; i < 8; ++i) {
      int row = mt * 16 + i + rsel;
      if (row < N_TOK) outb[row * CDIM + nt * 16 + lc] = acc[i] + pb;
    }
  }
}

// ---------------------------------------------------------------------------
extern "C" void kernel_launch(void* const* d_in, const int* in_sizes, int n_in,
                              void* d_out, int out_size, void* d_ws, size_t ws_size,
                              hipStream_t stream) {
  const float* x      = (const float*)d_in[0];
  const float* mask   = (const float*)d_in[1];
  const float* qkv_w  = (const float*)d_in[2];
  const float* qkv_b  = (const float*)d_in[3];
  const float* proj_w = (const float*)d_in[4];
  const float* proj_b = (const float*)d_in[5];
  const float* rpb    = (const float*)d_in[6];
  const int*   rpi    = (const int*)d_in[7];
  float* out = (float*)d_out;

  char* ws = (char*)d_ws;
  _Float16* w3h   = (_Float16*)ws;                        // 221184 B
  _Float16* pwh   = (_Float16*)(ws + 221184);             //  73728 B
  float*    biasF = (float*)(ws + 221184 + 73728);        // 230496 B

  const int total = 3 * CDIM * CDIM + CDIM * CDIM + NHEAD * N_TOK * N_TOK;
  prep_kernel<<<(total + 255) / 256, 256, 0, stream>>>(qkv_w, proj_w, rpb, rpi,
                                                       w3h, pwh, biasF);
  attn_kernel<<<NWINDOWS, 256, 0, stream>>>(x, mask, qkv_b, proj_b,
                                            w3h, pwh, biasF, out);
}
